// SymmetricTensorProduct_46497315947112
// MI455X (gfx1250) — matrix operand order repaired and checked by reading, not compiled, not run on hardware
//
#include <hip/hip_runtime.h>
#include <cstdint>
#include <cstddef>

// ---------------- problem constants (match reference) ----------------
#define S_IN   32
#define MUL    64
#define XDIM   (S_IN * MUL)        // 2048 floats per batch row
#define NP1    32
#define NP2    256
#define NP3    512
#define NPATH  (NP1 + NP2 + NP3)   // 800 = 25 * 32 (exact K-tiling, no padding)
#define KTILES (NPATH / 32)        // 25
#define ROWSTR 33                  // 32 x-values + dummy(=1.0); padded for LDS banks

typedef __attribute__((ext_vector_type(16))) _Float16 v16h;
typedef __attribute__((ext_vector_type(8)))  float    v8f;

// Dynamic-LDS layout:
//   [0,      51200) : bfrag — 25 ktiles x 2 ntiles x 32 lanes x 8 dwords (f16x2, WMMA-B order)
//   [51200,  68096) : xs    — 2 batch x 64 mul x 33 floats
//   [68096,  71296) : pidx  — 800 packed gather-index triples (i0|i1<<8|i2<<16, 32 = dummy 1.0)
#define BFRAG_BYTES 51200
#define XS_BYTES    16896
#define PIDX_BYTES  3200
#define SMEM_BYTES  (BFRAG_BYTES + XS_BYTES + PIDX_BYTES)

__device__ __forceinline__ uint32_t f16bits(float x) {
  _Float16 h = (_Float16)x;
  return (uint32_t)__builtin_bit_cast(unsigned short, h);
}

// flat path id k (0..799) -> (output slot, coefficient)
__device__ __forceinline__ void path_out_coef(
    int k,
    const int* __restrict__ out1, const float* __restrict__ c1,
    const int* __restrict__ out2, const float* __restrict__ c2,
    const int* __restrict__ out3, const float* __restrict__ c3,
    int& o, float& c) {
  if (k < NP1)            { o = out1[k];            c = c1[k]; }
  else if (k < NP1 + NP2) { int p = k - NP1;        o = out2[p]; c = c2[p]; }
  else                    { int p = k - NP1 - NP2;  o = out3[p]; c = c3[p]; }
}

// One K-tile (32 paths) of the row-tile GEMM, degree-specialized to skip dummy gathers.
// A layout (16-bit A 16x32): lane<16 holds row M=lane, K in {0..7,16..23};
// lane>=16 holds row M=lane-16, K in {8..15,24..31}; element e -> K = (e<8?e:e+8)+khalf.
template<int DEG>
__device__ __forceinline__ void contract_ktile(
    int t, const uint32_t* __restrict__ pidx, const float* __restrict__ xr,
    int khalf, const uint32_t* __restrict__ pb, v8f& acc0, v8f& acc1) {
  v16h a;
#pragma unroll
  for (int e = 0; e < 16; ++e) {
    const int kk = 32 * t + khalf + (e < 8 ? e : e + 8);
    const uint32_t u = pidx[kk];
    float g = xr[u & 255u];
    if (DEG >= 2) g *= xr[(u >> 8) & 255u];
    if (DEG >= 3) g *= xr[(u >> 16) & 255u];
    a[e] = (_Float16)g;
  }
  const v16h b0 = *(const v16h*)(pb + (size_t)t * 512);
  const v16h b1 = *(const v16h*)(pb + (size_t)t * 512 + 256);
  acc0 = __builtin_amdgcn_wmma_f32_16x16x32_f16(false, a, false, b0, (short)0, acc0, false, false);
  acc1 = __builtin_amdgcn_wmma_f32_16x16x32_f16(false, a, false, b1, (short)0, acc1, false, false);
}

__global__ __launch_bounds__(256) void stp_kernel(
    const float* __restrict__ x0,
    const int* __restrict__ in1, const int* __restrict__ out1, const float* __restrict__ c1,
    const int* __restrict__ in2, const int* __restrict__ out2, const float* __restrict__ c2,
    const int* __restrict__ in3, const int* __restrict__ out3, const float* __restrict__ c3,
    float* __restrict__ out) {
  extern __shared__ char smem[];
  uint32_t* bfrag = (uint32_t*)smem;                           // WMMA B fragments
  float*    xs    = (float*)(smem + BFRAG_BYTES);              // x tile
  uint32_t* pidx  = (uint32_t*)(smem + BFRAG_BYTES + XS_BYTES);// packed gather triples

  const int tid = threadIdx.x;
  const int blk = blockIdx.x;

  // ---- stage x tile: 2 batch rows of 2048 floats, coalesced global reads ----
  for (int t = tid; t < 2 * XDIM; t += 256) {
    const int bb = t >> 11, r = t & (XDIM - 1);
    const int i = r >> 6, m = r & 63;
    xs[(bb * 64 + m) * ROWSTR + i] = x0[(size_t)(blk * 2 + bb) * XDIM + r];
  }
  for (int t = tid; t < 128; t += 256) xs[t * ROWSTR + 32] = 1.0f;  // dummy multiplicand

  // ---- packed gather indices per path (unused slots -> 32 == 1.0) ----
  for (int k = tid; k < NPATH; k += 256) {
    uint32_t u;
    if (k < NP1) {
      u = (uint32_t)in1[k] | (32u << 8) | (32u << 16);
    } else if (k < NP1 + NP2) {
      const int p = k - NP1;
      u = (uint32_t)in2[2 * p] | ((uint32_t)in2[2 * p + 1] << 8) | (32u << 16);
    } else {
      const int p = k - NP1 - NP2;
      u = (uint32_t)in3[3 * p] | ((uint32_t)in3[3 * p + 1] << 8) |
          ((uint32_t)in3[3 * p + 2] << 16);
    }
    pidx[k] = u;
  }

  // ---- scatter matrix fragments in WMMA B-operand order ----
  // B is K(32) x N(16) f16: element e of v16h -> K = (e<8?e:e+8) + (lane<16 ? 0 : 8); N = lane&15.
  for (int idx = tid; idx < KTILES * 2 * 32 * 8; idx += 256) {
    const int r    = idx & 7;
    const int lane = (idx >> 3) & 31;
    const int nt   = (idx >> 8) & 1;
    const int t    = idx >> 9;
    const int n    = (lane & 15) + nt * 16;
    const int k0   = 32 * t + ((r & 4) ? 16 : 0) + 2 * (r & 3) + ((lane & 16) ? 8 : 0);
    int o; float c; uint32_t w = 0;
    path_out_coef(k0,     out1, c1, out2, c2, out3, c3, o, c);
    if (o == n) w |= f16bits(c);
    path_out_coef(k0 + 1, out1, c1, out2, c2, out3, c3, o, c);
    if (o == n) w |= f16bits(c) << 16;
    bfrag[idx] = w;
  }
  __syncthreads();

  // ---- per-wave 16x32 output tile: rows = 16 consecutive (b,m) channels ----
  const int wave = tid >> 5;
  const int lane = tid & 31;
  const int bloc = wave >> 2;           // 0..1 : which batch row of the block
  const int m0   = (wave & 3) * 16;     // channel-tile origin
  const int mrow = lane & 15;           // A-matrix row for this lane (both half-waves)
  const int khalf = (lane & 16) ? 8 : 0;
  const float* xr = &xs[(bloc * 64 + m0 + mrow) * ROWSTR];
  const uint32_t* pb = bfrag + lane * 8;

  v8f acc0 = {};
  v8f acc1 = {};
  // K-tiles split exactly on degree boundaries: 32 | 256 | 512 paths
  contract_ktile<1>(0, pidx, xr, khalf, pb, acc0, acc1);
  for (int t = 1; t < 9;  ++t) contract_ktile<2>(t, pidx, xr, khalf, pb, acc0, acc1);
  for (int t = 9; t < 25; ++t) contract_ktile<3>(t, pidx, xr, khalf, pb, acc0, acc1);

  // ---- epilogue: C/D layout lane<16 -> (M=v, N=lane); lane>=16 -> (M=v+8, N=lane-16)
  // out[b, o, m] flat = b*2048 + o*64 + m; 8 consecutive M = 8 consecutive m -> float4 x2
  const int b    = blk * 2 + bloc;
  const int n    = lane & 15;
  const int moff = m0 + ((lane & 16) ? 8 : 0);
  float* o0 = &out[(size_t)b * XDIM + (size_t)n * MUL + moff];
  float* o1 = &out[(size_t)b * XDIM + (size_t)(n + 16) * MUL + moff];
  *(float4*)(o0)     = make_float4(acc0[0], acc0[1], acc0[2], acc0[3]);
  *(float4*)(o0 + 4) = make_float4(acc0[4], acc0[5], acc0[6], acc0[7]);
  *(float4*)(o1)     = make_float4(acc1[0], acc1[1], acc1[2], acc1[3]);
  *(float4*)(o1 + 4) = make_float4(acc1[4], acc1[5], acc1[6], acc1[7]);
}

extern "C" void kernel_launch(void* const* d_in, const int* in_sizes, int n_in,
                              void* d_out, int out_size, void* d_ws, size_t ws_size,
                              hipStream_t stream) {
  const float* x0  = (const float*)d_in[0];
  const int*   in1 = (const int*)d_in[1];
  const int*   ou1 = (const int*)d_in[2];
  const float* c1  = (const float*)d_in[3];
  const int*   in2 = (const int*)d_in[4];
  const int*   ou2 = (const int*)d_in[5];
  const float* c2  = (const float*)d_in[6];
  const int*   in3 = (const int*)d_in[7];
  const int*   ou3 = (const int*)d_in[8];
  const float* c3  = (const float*)d_in[9];
  float* out = (float*)d_out;

  const int batch  = in_sizes[0] / XDIM;   // 4096
  const int blocks = batch / 2;            // 2 batch rows per block

  stp_kernel<<<dim3(blocks), dim3(256), SMEM_BYTES, stream>>>(
      x0, in1, ou1, c1, in2, ou2, c2, in3, ou3, c3, out);
}
